// SpareNetEncode_22419729285790
// MI455X (gfx1250) — matrix operand order repaired and checked by reading, not compiled, not run on hardware
//
#include <hip/hip_runtime.h>
#include <hip/hip_bf16.h>
#include <stdint.h>

typedef __attribute__((ext_vector_type(16))) _Float16 v16h;
typedef __attribute__((ext_vector_type(4)))  _Float16 v4h;
typedef __attribute__((ext_vector_type(8)))  float    v8f;

#define BATCH 8
#define NPTS  2048
#define NSH   11            // log2(NPTS)
#define KNN   8
#define TM 128              // output rows per block (8 waves x 16)
#define TN 64               // output cols per block (4 subtiles x 16)
#define TKK 32              // K step
#define NTHREADS 256

#define A_FRAGS (TM / 16)
#define B_FRAGS (TN / 16)
#define A_LDS (A_FRAGS * 32 * 16)   // 4096 halves
#define B_LDS (B_FRAGS * 32 * 16)   // 2048 halves

// ---- stage one K-tile into fragment-swizzled LDS (CDNA5 16x16x32 f16 layout) ----
// A: (row,k) -> lane=(row&15)+16*((k>>3)&1), e=(k&7)+8*(k>>4)
// B: (k,col) -> lane=(col&15)+16*((k>>3)&1), e=(k&7)+8*(k>>4)
template <bool GRAM, bool CHECK>
__device__ __forceinline__ void stage_tiles(
    _Float16* __restrict__ As, _Float16* __restrict__ Bs,
    const float* __restrict__ A, int lda, const float* __restrict__ Xb,
    int Cdim, int o0, int n0, int k0) {
#pragma unroll
  for (int u = 0; u < (TM * TKK / 4) / NTHREADS; ++u) {
    const int idx  = u * NTHREADS + (int)threadIdx.x;
    const int row  = idx & (TM - 1);
    const int rest = idx >> 7;              // 0..7
    const int g = rest & 1, kh = (rest >> 1) & 1, q = rest >> 2;
    const int kbase = 8 * g + 16 * kh + 4 * q;
    const int oo = o0 + row;
    v4h chunk;
#pragma unroll
    for (int j = 0; j < 4; ++j) {
      const int cc = k0 + kbase + j;
      float v;
      if (CHECK && cc >= Cdim) v = 0.f;
      else v = GRAM ? Xb[(long)cc * NPTS + oo] : A[(long)oo * lda + cc];
      chunk[j] = (_Float16)v;
    }
    const int slane = (row & 15) + 16 * g;
    *(v4h*)(&As[((row >> 4) * 32 + slane) * 16 + 4 * q + 8 * kh]) = chunk;
  }
#pragma unroll
  for (int u = 0; u < (TKK * TN / 4) / NTHREADS; ++u) {
    const int idx  = u * NTHREADS + (int)threadIdx.x;
    const int col  = idx & (TN - 1);
    const int rest = idx >> 6;              // 0..7
    const int g = rest & 1, c4 = rest >> 1;
    const int kbase = 8 * g + 16 * (c4 >> 1) + 4 * (c4 & 1);
    v4h chunk;
#pragma unroll
    for (int j = 0; j < 4; ++j) {
      const int cc = k0 + kbase + j;
      float v;
      if (CHECK && cc >= Cdim) v = 0.f;
      else v = Xb[(long)cc * NPTS + (n0 + col)];
      chunk[j] = (_Float16)v;
    }
    const int slane = 16 * g + (col & 15);
    *(v4h*)(&Bs[((col >> 4) * 32 + slane) * 16 + 4 * c4]) = chunk;
  }
}

// ---- tiled WMMA GEMM with ping-pong LDS: Y[b,o,n] = sum_c A[o,c]*X[b,c,n] ----
// GRAM: A tile is X^T of the same batch (Gram matrix x^T x). Grids exactly cover O/N.
template <bool GRAM>
__global__ void __launch_bounds__(NTHREADS)
wmma_gemm_kernel(const float* __restrict__ A, int lda,
                 const float* __restrict__ X, long xBatchStride,
                 float* __restrict__ Y, long yBatchStride, int Cdim) {
  __shared__ __align__(32) _Float16 As[2][A_LDS];
  __shared__ __align__(32) _Float16 Bs[2][B_LDS];
  const int b    = blockIdx.z;
  const int n0   = blockIdx.x * TN;
  const int o0   = blockIdx.y * TM;
  const int wave = threadIdx.x >> 5;
  const int lane = threadIdx.x & 31;
  const float* Xb = X + (long)b * xBatchStride;

  v8f zero = {0.f, 0.f, 0.f, 0.f, 0.f, 0.f, 0.f, 0.f};
  v8f acc[4];
#pragma unroll
  for (int t = 0; t < 4; ++t) acc[t] = zero;

  const int nk = (Cdim + TKK - 1) / TKK;
  if (Cdim >= TKK) stage_tiles<GRAM, false>(As[0], Bs[0], A, lda, Xb, Cdim, o0, n0, 0);
  else             stage_tiles<GRAM, true >(As[0], Bs[0], A, lda, Xb, Cdim, o0, n0, 0);
  __syncthreads();

  int cur = 0;
  for (int it = 1; it < nk; ++it) {
    // stage next tile first (global loads in flight during WMMA); always a full tile
    stage_tiles<GRAM, false>(As[cur ^ 1], Bs[cur ^ 1], A, lda, Xb, Cdim, o0, n0, it * TKK);
    const v16h af = *(const v16h*)(&As[cur][(wave * 32 + lane) * 16]);
#pragma unroll
    for (int t = 0; t < 4; ++t) {
      const v16h bf = *(const v16h*)(&Bs[cur][(t * 32 + lane) * 16]);
      acc[t] = __builtin_amdgcn_wmma_f32_16x16x32_f16(
          false, af, false, bf, (short)0, acc[t], false, false);
    }
    __syncthreads();
    cur ^= 1;
  }
  {
    const v16h af = *(const v16h*)(&As[cur][(wave * 32 + lane) * 16]);
#pragma unroll
    for (int t = 0; t < 4; ++t) {
      const v16h bf = *(const v16h*)(&Bs[cur][(t * 32 + lane) * 16]);
      acc[t] = __builtin_amdgcn_wmma_f32_16x16x32_f16(
          false, af, false, bf, (short)0, acc[t], false, false);
    }
  }

  // C/D layout: VGPR r -> row = r + 8*(lane>=16), col = lane&15
  float* Yb = Y + (long)b * yBatchStride;
  const int ccol = lane & 15;
  const int rowBase = o0 + wave * 16 + (lane >> 4) * 8;
#pragma unroll
  for (int t = 0; t < 4; ++t) {
    const int colBase = n0 + t * 16 + ccol;
#pragma unroll
    for (int r = 0; r < 8; ++r)
      Yb[(long)(rowBase + r) * NPTS + colBase] = acc[t][r];
  }
}

// ---------------- small support kernels ----------------

__global__ void norms_kernel(const float* __restrict__ x, long bStride, int C,
                             float* __restrict__ xx) {
  const int t = blockIdx.x * blockDim.x + threadIdx.x;
  if (t >= BATCH * NPTS) return;
  const int b = t >> NSH, n = t & (NPTS - 1);
  const float* xb = x + (long)b * bStride;
  float s = 0.f;
  for (int c = 0; c < C; ++c) { const float v = xb[(long)c * NPTS + n]; s += v * v; }
  xx[t] = s;
}

// top-8 of score(m) = 2*gram[n][m] - xx[m]  (the -xx[n] term is row-constant)
__global__ void topk_kernel(const float* __restrict__ gram, const float* __restrict__ xx,
                            int* __restrict__ idx) {
  const int t = blockIdx.x * blockDim.x + threadIdx.x;
  if (t >= BATCH * NPTS) return;
  const int b = t >> NSH;
  const float4* row4 = (const float4*)(gram + (long)t * NPTS);
  const float4* xb4  = (const float4*)(xx + ((long)b << NSH));
  float bv[KNN]; int bi[KNN];
#pragma unroll
  for (int i = 0; i < KNN; ++i) { bv[i] = -3.4e38f; bi[i] = 0; }
  for (int m4 = 0; m4 < NPTS / 4; ++m4) {
    const float4 gv = row4[m4];
    const float4 xv = xb4[m4];
    float s4[4] = {2.f * gv.x - xv.x, 2.f * gv.y - xv.y,
                   2.f * gv.z - xv.z, 2.f * gv.w - xv.w};
#pragma unroll
    for (int j = 0; j < 4; ++j) {
      const float s = s4[j];
      if (s > bv[KNN - 1]) {
        int p = KNN - 1;
#pragma unroll
        for (int i = KNN - 2; i >= 0; --i)
          if (s > bv[i]) { bv[i + 1] = bv[i]; bi[i + 1] = bi[i]; p = i; }
        bv[p] = s; bi[p] = 4 * m4 + j;
      }
    }
  }
#pragma unroll
  for (int i = 0; i < KNN; ++i) idx[(long)t * KNN + i] = bi[i];
}

__global__ void wdiff_kernel(const float* __restrict__ w, float* __restrict__ wd, int O, int C) {
  const int t = blockIdx.x * blockDim.x + threadIdx.x;
  if (t >= O * C) return;
  const int o = t / C, c = t % C;
  wd[t] = w[(long)o * 2 * C + C + c] - w[(long)o * 2 * C + c];
}

// per-channel BN stats of the (gathered) edge-conv pre-activation
__global__ void bn_stats_edge_kernel(const float* __restrict__ g, const float* __restrict__ y,
                                     const int* __restrict__ idx, int O,
                                     float* __restrict__ mean, float* __restrict__ rstd) {
  const int o = blockIdx.x;
  float s1 = 0.f, s2 = 0.f;
  for (int t = threadIdx.x; t < BATCH * NPTS; t += blockDim.x) {
    const int b = t >> NSH, n = t & (NPTS - 1);
    const float* gb = g + ((long)b * O + o) * NPTS;
    const float yv = y[((long)b * O + o) * NPTS + n];
    const int* id = idx + ((long)t) * KNN;
#pragma unroll
    for (int k = 0; k < KNN; ++k) { const float v = gb[id[k]] + yv; s1 += v; s2 += v * v; }
  }
  __shared__ float r1[256], r2[256];
  r1[threadIdx.x] = s1; r2[threadIdx.x] = s2; __syncthreads();
  for (int st = 128; st > 0; st >>= 1) {
    if ((int)threadIdx.x < st) { r1[threadIdx.x] += r1[threadIdx.x + st]; r2[threadIdx.x] += r2[threadIdx.x + st]; }
    __syncthreads();
  }
  if (threadIdx.x == 0) {
    const float cnt = (float)BATCH * NPTS * KNN;
    const float m = r1[0] / cnt;
    const float var = r2[0] / cnt - m * m;
    mean[o] = m;
    rstd[o] = rsqrtf(var + 1e-5f);
  }
}

// fused gather + BN + leakyReLU + max-over-k (+ residual); O is a power of two
__global__ void edge_out_kernel(const float* __restrict__ g, const float* __restrict__ y,
                                const float* __restrict__ r, const int* __restrict__ idx,
                                const float* __restrict__ gamma, const float* __restrict__ beta,
                                const float* __restrict__ mean, const float* __restrict__ rstd,
                                int O, int oShift, float* __restrict__ out, long outBatchStride) {
  const int t = blockIdx.x * blockDim.x + threadIdx.x;
  if (t >= (BATCH << (NSH + oShift))) return;
  const int n = t & (NPTS - 1);
  const int rest = t >> NSH;
  const int o = rest & (O - 1);
  const int b = rest >> oShift;
  const float* gb = g + ((long)b * O + o) * NPTS;
  const float yv = y[((long)b * O + o) * NPTS + n];
  const int* id = idx + ((long)((b << NSH) + n)) * KNN;
  const float ga = gamma[o], be = beta[o], m = mean[o], rs = rstd[o];
  float mx = -3.4e38f;
#pragma unroll
  for (int k = 0; k < KNN; ++k) {
    const float v = gb[id[k]] + yv;
    float tt = ga * (v - m) * rs + be;
    tt = tt > 0.f ? tt : 0.2f * tt;
    mx = fmaxf(mx, tt);
  }
  if (r) mx += r[((long)b * O + o) * NPTS + n];
  out[(long)b * outBatchStride + (long)o * NPTS + n] = mx;
}

__global__ void bn_stats_plain_kernel(const float* __restrict__ y, int O,
                                      float* __restrict__ mean, float* __restrict__ rstd) {
  const int o = blockIdx.x;
  float s1 = 0.f, s2 = 0.f;
  for (int t = threadIdx.x; t < BATCH * NPTS; t += blockDim.x) {
    const int b = t >> NSH, n = t & (NPTS - 1);
    const float v = y[((long)b * O + o) * NPTS + n];
    s1 += v; s2 += v * v;
  }
  __shared__ float r1[256], r2[256];
  r1[threadIdx.x] = s1; r2[threadIdx.x] = s2; __syncthreads();
  for (int st = 128; st > 0; st >>= 1) {
    if ((int)threadIdx.x < st) { r1[threadIdx.x] += r1[threadIdx.x + st]; r2[threadIdx.x] += r2[threadIdx.x + st]; }
    __syncthreads();
  }
  if (threadIdx.x == 0) {
    const float cnt = (float)BATCH * NPTS;
    const float m = r1[0] / cnt;
    const float var = r2[0] / cnt - m * m;
    mean[o] = m;
    rstd[o] = rsqrtf(var + 1e-5f);
  }
}

// farthest point sampling: one block per batch, points resident in LDS
__global__ void fps_kernel(const float* __restrict__ coor, long bStride, int N, int m,
                           int* __restrict__ sel, float* __restrict__ coorOut, long coBStride) {
  __shared__ float px[NPTS], py[NPTS], pz[NPTS], dist[NPTS];
  __shared__ float rv[256];
  __shared__ int   ri[256];
  __shared__ int   selS[512];
  const int b = blockIdx.x;
  const float* cb = coor + (long)b * bStride;
  for (int i = threadIdx.x; i < N; i += blockDim.x) {
    px[i] = cb[i]; py[i] = cb[N + i]; pz[i] = cb[2 * N + i];
    dist[i] = 1e10f;
  }
  __syncthreads();
  int last = 0;
  for (int it = 0; it < m; ++it) {
    if (threadIdx.x == 0) { sel[b * m + it] = last; selS[it] = last; }
    const float lx = px[last], ly = py[last], lz = pz[last];
    float bestv = -1.f; int besti = 0;
    for (int i = threadIdx.x; i < N; i += blockDim.x) {
      const float dx = px[i] - lx, dy = py[i] - ly, dz = pz[i] - lz;
      const float d = dx * dx + dy * dy + dz * dz;
      const float dd = fminf(dist[i], d);
      dist[i] = dd;
      if (dd > bestv) { bestv = dd; besti = i; }
    }
    rv[threadIdx.x] = bestv; ri[threadIdx.x] = besti;
    __syncthreads();
    for (int st = 128; st > 0; st >>= 1) {
      if ((int)threadIdx.x < st && rv[threadIdx.x + st] > rv[threadIdx.x]) {
        rv[threadIdx.x] = rv[threadIdx.x + st]; ri[threadIdx.x] = ri[threadIdx.x + st];
      }
      __syncthreads();
    }
    last = ri[0];
    __syncthreads();
  }
  if (coorOut) {
    for (int j = threadIdx.x; j < m; j += blockDim.x) {
      const int s = selS[j];
      coorOut[(long)b * coBStride + 0 * m + j] = px[s];
      coorOut[(long)b * coBStride + 1 * m + j] = py[s];
      coorOut[(long)b * coBStride + 2 * m + j] = pz[s];
    }
  }
}

// compose the 3 FPS index stages; emit final coordinates directly to d_out
__global__ void compose_gather_kernel(const int* __restrict__ s1, const int* __restrict__ s2,
                                      const int* __restrict__ s3, int* __restrict__ selF,
                                      const float* __restrict__ x, float* __restrict__ outCoor) {
  const int b = blockIdx.x, j = threadIdx.x;  // 128 threads
  const int sf = s1[b * 512 + s2[b * 256 + s3[b * 128 + j]]];
  selF[b * 128 + j] = sf;
#pragma unroll
  for (int c = 0; c < 3; ++c)
    outCoor[b * 384 + c * 128 + j] = x[((long)b * 3 + c) * NPTS + sf];
}

// on-the-fly BN+leaky at the 128 sampled points, then max & mean pool
__global__ void final_pool_kernel(const float* __restrict__ y5, const int* __restrict__ selF,
                                  const float* __restrict__ g5, const float* __restrict__ b5,
                                  const float* __restrict__ mean5, const float* __restrict__ rstd5,
                                  float* __restrict__ outF) {
  const int t = blockIdx.x * blockDim.x + threadIdx.x;
  if (t >= BATCH * 2048) return;
  const int b = t >> 11, o = t & 2047;
  const float* yb = y5 + ((long)b * 2048 + o) * NPTS;
  const int* sb = selF + b * 128;
  const float ga = g5[o], be = b5[o], m = mean5[o], rs = rstd5[o];
  float mx = -3.4e38f, sum = 0.f;
  for (int j = 0; j < 128; ++j) {
    float v = ga * (yb[sb[j]] - m) * rs + be;
    v = v > 0.f ? v : 0.2f * v;
    mx = fmaxf(mx, v);
    sum += v;
  }
  outF[b * 4096 + o] = mx;
  outF[b * 4096 + 2048 + o] = sum * (1.f / 128.f);
}

// ---------------- host orchestration ----------------

static inline int ilog2(int v) { int s = 0; while ((1 << s) < v) ++s; return s; }

extern "C" void kernel_launch(void* const* d_in, const int* in_sizes, int n_in,
                              void* d_out, int out_size, void* d_ws, size_t ws_size,
                              hipStream_t stream) {
  (void)in_sizes; (void)n_in; (void)out_size; (void)ws_size;
  const int N = NPTS, B = BATCH;

  const float* x   = (const float*)d_in[0];
  const float* w1  = (const float*)d_in[1];
  const float* w2  = (const float*)d_in[2];
  const float* w3  = (const float*)d_in[3];
  const float* w4  = (const float*)d_in[4];
  const float* w5  = (const float*)d_in[5];
  const float* rw1 = (const float*)d_in[6];
  const float* rw2 = (const float*)d_in[7];
  const float* rw3 = (const float*)d_in[8];
  const float* gs[5] = {(const float*)d_in[9],  (const float*)d_in[11],
                        (const float*)d_in[13], (const float*)d_in[15],
                        (const float*)d_in[17]};
  const float* bs[5] = {(const float*)d_in[10], (const float*)d_in[12],
                        (const float*)d_in[14], (const float*)d_in[16],
                        (const float*)d_in[18]};
  float* out = (float*)d_out;

  // bump allocator over d_ws
  char* base = (char*)d_ws;
  size_t off = 0;
  auto alloc = [&](size_t bytes) -> char* {
    char* p = base + off;
    off += (bytes + 255) & ~(size_t)255;
    return p;
  };
  float* gram  = (float*)alloc((size_t)B * N * N * 4);        // 128 MB
  float* y5    = (float*)alloc((size_t)B * 2048 * N * 4);     // 128 MB
  float* hbuf  = (float*)alloc((size_t)B * 1024 * N * 4);     // 64 MB
  float* gbuf  = (float*)alloc((size_t)B * 512 * N * 4);      // 32 MB
  float* ybuf  = (float*)alloc((size_t)B * 512 * N * 4);      // 32 MB
  float* rbuf  = (float*)alloc((size_t)B * 512 * N * 4);      // 32 MB
  float* norms = (float*)alloc((size_t)B * N * 4);
  int*   idxb  = (int*)alloc((size_t)B * N * KNN * 4);
  float* wdif  = (float*)alloc((size_t)512 * 512 * 4);
  float* meanE = (float*)alloc(2048 * 4);
  float* rstdE = (float*)alloc(2048 * 4);
  float* mean5 = (float*)alloc(2048 * 4);
  float* rstd5 = (float*)alloc(2048 * 4);
  int*   sel1  = (int*)alloc((size_t)B * 512 * 4);
  int*   sel2  = (int*)alloc((size_t)B * 256 * 4);
  int*   sel3  = (int*)alloc((size_t)B * 128 * 4);
  int*   selF  = (int*)alloc((size_t)B * 128 * 4);
  float* coor1 = (float*)alloc((size_t)B * 3 * 512 * 4);
  float* coor2 = (float*)alloc((size_t)B * 3 * 256 * 4);

  const long hStride = (long)1024 * N;

  auto edge_stage = [&](const float* xin, long xinStride, int Cin,
                        const float* w, const float* rw,
                        const float* gamma, const float* beta,
                        int O, float* outp) {
    norms_kernel<<<(B * N + 255) / 256, 256, 0, stream>>>(xin, xinStride, Cin, norms);
    dim3 gg(N / TN, N / TM, B);
    wmma_gemm_kernel<true><<<gg, NTHREADS, 0, stream>>>(xin, 0, xin, xinStride,
                                                        gram, (long)N * N, Cin);
    topk_kernel<<<(B * N + 127) / 128, 128, 0, stream>>>(gram, norms, idxb);
    wdiff_kernel<<<(O * Cin + 255) / 256, 256, 0, stream>>>(w, wdif, O, Cin);
    dim3 gc(N / TN, O / TM, B);
    wmma_gemm_kernel<false><<<gc, NTHREADS, 0, stream>>>(w, 2 * Cin, xin, xinStride,
                                                         gbuf, (long)O * N, Cin);
    wmma_gemm_kernel<false><<<gc, NTHREADS, 0, stream>>>(wdif, Cin, xin, xinStride,
                                                         ybuf, (long)O * N, Cin);
    if (rw)
      wmma_gemm_kernel<false><<<gc, NTHREADS, 0, stream>>>(rw, Cin, xin, xinStride,
                                                           rbuf, (long)O * N, Cin);
    bn_stats_edge_kernel<<<O, 256, 0, stream>>>(gbuf, ybuf, idxb, O, meanE, rstdE);
    const int tot = B * O * N;
    edge_out_kernel<<<(tot + 255) / 256, 256, 0, stream>>>(
        gbuf, ybuf, rw ? rbuf : nullptr, idxb, gamma, beta, meanE, rstdE,
        O, ilog2(O), outp, hStride);
  };

  // x1..x4 written channel-contiguously into hbuf => concat is free
  edge_stage(x,                (long)3 * N,  3,   w1, nullptr, gs[0], bs[0], 128, hbuf);
  edge_stage(hbuf,             hStride,      128, w2, rw1,     gs[1], bs[1], 128, hbuf + 128 * N);
  edge_stage(hbuf + 128 * N,   hStride,      128, w3, rw2,     gs[2], bs[2], 256, hbuf + 256 * N);
  edge_stage(hbuf + 256 * N,   hStride,      256, w4, rw3,     gs[3], bs[3], 512, hbuf + 512 * N);

  // y5 = w5 (2048x1024) . h
  {
    dim3 g5g(N / TN, 2048 / TM, B);
    wmma_gemm_kernel<false><<<g5g, NTHREADS, 0, stream>>>(w5, 1024, hbuf, hStride,
                                                          y5, (long)2048 * N, 1024);
    bn_stats_plain_kernel<<<2048, 256, 0, stream>>>(y5, 2048, mean5, rstd5);
  }

  // FPS chain on coordinates only
  fps_kernel<<<B, 256, 0, stream>>>(x,     (long)3 * N,   N,   512, sel1, coor1, (long)3 * 512);
  fps_kernel<<<B, 256, 0, stream>>>(coor1, (long)3 * 512, 512, 256, sel2, coor2, (long)3 * 256);
  fps_kernel<<<B, 256, 0, stream>>>(coor2, (long)3 * 256, 256, 128, sel3, nullptr, 0);

  compose_gather_kernel<<<B, 128, 0, stream>>>(sel1, sel2, sel3, selF, x, out);
  final_pool_kernel<<<(B * 2048 + 255) / 256, 256, 0, stream>>>(
      y5, selF, gs[4], bs[4], mean5, rstd5, out + B * 3 * 128);
}
